// MyCustomLayer_5626407158334
// MI455X (gfx1250) — matrix operand order repaired and checked by reading, not compiled, not run on hardware
//
#include <hip/hip_runtime.h>
#include <hip/hip_bf16.h>

// CDNA5 / gfx1250, wave32 WMMA implementation of the pitchfork-LSTM.
// Phase 1: xz = x@W + b as one big f16 WMMA GEMM (f32 accumulate).
// Phase 2: 512 per-timestep kernels: z = xz_t + h@U (WMMA), gates, c/h update.

typedef __attribute__((ext_vector_type(16))) _Float16 v16h;
typedef __attribute__((ext_vector_type(8)))  _Float16 v8h;
typedef __attribute__((ext_vector_type(8)))  float    v8f;

#define B_ 64
#define T_ 512
#define D_ 256
#define H_ 512
#define G_ 2048   // 4*H

__device__ __forceinline__ float sigmoidf_(float x) { return 1.0f / (1.0f + __expf(-x)); }
__device__ __forceinline__ float pitchfork_(float x) { return x * x * x - 2.0f * x; }

// ---------------- conversion / init kernels ----------------

__global__ void __launch_bounds__(256)
k_zero_state(_Float16* __restrict__ h0, float* __restrict__ c) {
  int i = blockIdx.x * blockDim.x + threadIdx.x;
  if (i < B_ * H_) { h0[i] = (_Float16)0.0f; c[i] = 0.0f; }
}

// out[outRows][outCols] = transpose of in[outCols][outRows], f32 -> f16
__global__ void __launch_bounds__(256)
k_transpose_f16(const float* __restrict__ in, _Float16* __restrict__ out,
                int outRows, int outCols) {
  int idx = blockIdx.x * blockDim.x + threadIdx.x;
  if (idx >= outRows * outCols) return;
  int j = idx % outCols;           // inner (K) index of out row
  int i = idx / outCols;           // out row
  out[idx] = (_Float16)in[(size_t)j * outRows + i];
}

// x[B][T][D] f32 -> x16[T][B][D] f16
__global__ void __launch_bounds__(256)
k_permute_x(const float* __restrict__ x, _Float16* __restrict__ out) {
  int idx = blockIdx.x * blockDim.x + threadIdx.x;   // over T*B*D
  if (idx >= T_ * B_ * D_) return;
  int d = idx % D_;
  int rest = idx / D_;             // rest = t*B + b
  int b = rest % B_;
  int t = rest / B_;
  out[idx] = (_Float16)x[((size_t)b * T_ + t) * D_ + d];
}

// ---------------- WMMA fragment helpers ----------------
// A (16x32 f16, MxK): lane<16 -> row=lane, K=k0..k0+15
//                     lane>=16 -> row=lane-16, K=k0+16..k0+31
// B (32x16 f16, KxN): lane holds column N=lane%16; lane<16 -> K=k0..k0+15,
//                     lane>=16 -> K=k0+16..k0+31 (contiguous 32B when K-major).

__device__ __forceinline__ v16h load_a_frag(const _Float16* arow, int k0) {
  v8h lo = *(const v8h*)(arow + k0);
  v8h hi = *(const v8h*)(arow + k0 + 8);
  v16h a;
#pragma unroll
  for (int i = 0; i < 8; ++i) { a[i] = lo[i]; a[i + 8] = hi[i]; }
  return a;
}

// ---------------- phase 1: xz = x@W + b ----------------
// X   : [T*B, D] f16 row-major
// Wt  : [G, D]   f16 row-major (transposed W, so columns of W are contiguous)
// XZ  : [T*B, G] f16
__global__ void __launch_bounds__(256)
k_xz_gemm(const _Float16* __restrict__ X, const _Float16* __restrict__ Wt,
          const float* __restrict__ bias, _Float16* __restrict__ XZ) {
  const int wave = threadIdx.x >> 5;
  const int lane = threadIdx.x & 31;
  const int w = blockIdx.x * 8 + wave;       // 65536 waves total
  const int nstrip = w & 31;                 // 32 strips of 64 columns
  const int mtile  = w >> 5;                 // 2048 row tiles
  const int m0 = mtile * 16;
  const int n0 = nstrip * 64;
  const int lm = lane & 15;
  const bool hiL = lane >= 16;
  const _Float16* arow = X + (size_t)(m0 + lm) * D_ + (hiL ? 16 : 0);
  const int bk = hiL ? 16 : 0;
  const _Float16* brow0 = Wt + (size_t)(n0 + lm) * D_ + bk;

  v8f acc0 = {}, acc1 = {}, acc2 = {}, acc3 = {};
#pragma unroll
  for (int k0 = 0; k0 < D_; k0 += 32) {
    if (k0 + 64 < D_) __builtin_prefetch(arow + k0 + 64, 0, 1);
    v16h a  = load_a_frag(arow, k0);
    v16h b0 = *(const v16h*)(brow0 + k0);
    v16h b1 = *(const v16h*)(brow0 + (size_t)16 * D_ + k0);
    v16h b2 = *(const v16h*)(brow0 + (size_t)32 * D_ + k0);
    v16h b3 = *(const v16h*)(brow0 + (size_t)48 * D_ + k0);
    acc0 = __builtin_amdgcn_wmma_f32_16x16x32_f16(false, a, false, b0, (short)0, acc0, false, false);
    acc1 = __builtin_amdgcn_wmma_f32_16x16x32_f16(false, a, false, b1, (short)0, acc1, false, false);
    acc2 = __builtin_amdgcn_wmma_f32_16x16x32_f16(false, a, false, b2, (short)0, acc2, false, false);
    acc3 = __builtin_amdgcn_wmma_f32_16x16x32_f16(false, a, false, b3, (short)0, acc3, false, false);
  }

  // C/D layout: VGPR r: lanes 0-15 -> M=r, N=lane; lanes 16-31 -> M=r+8, N=lane-16
  const int mbase = m0 + (hiL ? 8 : 0);
  const int c0 = n0 + lm;
#pragma unroll
  for (int r = 0; r < 8; ++r) {
    size_t row = (size_t)(mbase + r) * G_;
    XZ[row + c0     ] = (_Float16)(acc0[r] + bias[c0     ]);
    XZ[row + c0 + 16] = (_Float16)(acc1[r] + bias[c0 + 16]);
    XZ[row + c0 + 32] = (_Float16)(acc2[r] + bias[c0 + 32]);
    XZ[row + c0 + 48] = (_Float16)(acc3[r] + bias[c0 + 48]);
  }
}

// ---------------- phase 2: one recurrent timestep ----------------
// XZt : xz slice for step t: [B, G] f16
// Ut  : [G, H] f16 (transposed U)
// Hin : [B, H] f16 ; Hout: [B, H] f16 ; Cbuf: [B, H] f32 (updated in place)
// HoutF32: [B, H] f32 (d_out; last step leaves the answer)
__global__ void __launch_bounds__(256)
k_step(const _Float16* __restrict__ XZt, const _Float16* __restrict__ Ut,
       const _Float16* __restrict__ Hin, _Float16* __restrict__ Hout,
       float* __restrict__ Cbuf, float* __restrict__ HoutF32) {
  const int wave = threadIdx.x >> 5;
  const int lane = threadIdx.x & 31;
  const int w = blockIdx.x * 8 + wave;       // 128 waves total
  const int ntile = w & 31;                  // 32 tiles across H
  const int mtile = w >> 5;                  // 4 tiles across B
  const int m0 = mtile * 16;
  const int n0 = ntile * 16;
  const int lm = lane & 15;
  const bool hiL = lane >= 16;
  const _Float16* arow = Hin + (size_t)(m0 + lm) * H_ + (hiL ? 16 : 0);
  const int bk = hiL ? 16 : 0;

  v8f acc[4] = {{}, {}, {}, {}};
  for (int k0 = 0; k0 < H_; k0 += 32) {
    if (k0 + 64 < H_) __builtin_prefetch(arow + k0 + 64, 0, 1);
    v16h a = load_a_frag(arow, k0);
#pragma unroll
    for (int gte = 0; gte < 4; ++gte) {
      const v16h b = *(const v16h*)(Ut + (size_t)(gte * H_ + n0 + lm) * H_ + k0 + bk);
      acc[gte] = __builtin_amdgcn_wmma_f32_16x16x32_f16(false, a, false, b, (short)0, acc[gte], false, false);
    }
  }

  const int mbase = m0 + (hiL ? 8 : 0);
  const int n = n0 + lm;
#pragma unroll
  for (int r = 0; r < 8; ++r) {
    const int m = mbase + r;
    const size_t zrow = (size_t)m * G_;
    const float zi = acc[0][r] + (float)XZt[zrow + n          ];
    const float zf = acc[1][r] + (float)XZt[zrow + n + H_     ];
    const float zg = acc[2][r] + (float)XZt[zrow + n + 2 * H_ ];
    const float zo = acc[3][r] + (float)XZt[zrow + n + 3 * H_ ];
    const float ig = sigmoidf_(zi);
    const float fg = sigmoidf_(zf);
    const float gg = pitchfork_(zg);
    const float og = sigmoidf_(zo);
    const size_t ci = (size_t)m * H_ + n;
    const float c = fg * Cbuf[ci] + ig * gg;
    Cbuf[ci] = c;
    const float h = og * pitchfork_(c);
    Hout[ci] = (_Float16)h;
    HoutF32[ci] = h;
  }
}

// ---------------- launch ----------------

extern "C" void kernel_launch(void* const* d_in, const int* in_sizes, int n_in,
                              void* d_out, int out_size, void* d_ws, size_t ws_size,
                              hipStream_t stream) {
  const float* x    = (const float*)d_in[0];   // [B,T,D]
  const float* W    = (const float*)d_in[1];   // [D,G]
  const float* U    = (const float*)d_in[2];   // [H,G]
  const float* bias = (const float*)d_in[3];   // [G]
  float* out = (float*)d_out;                  // [B,H]

  char* ws = (char*)d_ws;
  const size_t XZ_BYTES = (size_t)T_ * B_ * G_ * sizeof(_Float16);   // 128 MB
  const size_t X_BYTES  = (size_t)T_ * B_ * D_ * sizeof(_Float16);   // 16 MB
  const size_t WT_BYTES = (size_t)G_ * D_ * sizeof(_Float16);        // 1 MB
  const size_t UT_BYTES = (size_t)G_ * H_ * sizeof(_Float16);        // 2 MB
  const size_t H_BYTES  = (size_t)B_ * H_ * sizeof(_Float16);        // 64 KB

  _Float16* xz16 = (_Float16*)(ws);
  _Float16* x16  = (_Float16*)(ws + XZ_BYTES);
  _Float16* wt   = (_Float16*)(ws + XZ_BYTES + X_BYTES);
  _Float16* ut   = (_Float16*)(ws + XZ_BYTES + X_BYTES + WT_BYTES);
  _Float16* h0   = (_Float16*)(ws + XZ_BYTES + X_BYTES + WT_BYTES + UT_BYTES);
  _Float16* h1   = (_Float16*)(ws + XZ_BYTES + X_BYTES + WT_BYTES + UT_BYTES + H_BYTES);
  float*    cbuf = (float*)   (ws + XZ_BYTES + X_BYTES + WT_BYTES + UT_BYTES + 2 * H_BYTES);

  // conversions + init
  k_permute_x<<<(T_ * B_ * D_) / 256, 256, 0, stream>>>(x, x16);
  k_transpose_f16<<<(G_ * D_) / 256, 256, 0, stream>>>(W, wt, G_, D_);
  k_transpose_f16<<<(G_ * H_) / 256, 256, 0, stream>>>(U, ut, G_, H_);
  k_zero_state<<<(B_ * H_ + 255) / 256, 256, 0, stream>>>(h0, cbuf);

  // phase 1: xz = x@W + b  (2048 row tiles x 32 col strips = 65536 waves)
  k_xz_gemm<<<8192, 256, 0, stream>>>(x16, wt, bias, xz16);

  // phase 2: sequential scan, one kernel per timestep (128 waves each)
  _Float16* hin = h0;
  _Float16* hout = h1;
  for (int t = 0; t < T_; ++t) {
    k_step<<<16, 256, 0, stream>>>(xz16 + (size_t)t * B_ * G_, ut, hin, hout, cbuf, out);
    _Float16* tmp = hin; hin = hout; hout = tmp;
  }
}
